// HyenaBaseline_73847667687715
// MI455X (gfx1250) — matrix operand order repaired and checked by reading, not compiled, not run on hardware
//
#include <hip/hip_runtime.h>
#include <hip/hip_bf16.h>

// ---------------------------------------------------------------------------
// Hyena forward for MI455X (gfx1250, wave32, WMMA).
// GEMMs: v_wmma_f32_16x16x32_bf16, fp32 accumulate, async global->LDS staging.
// ---------------------------------------------------------------------------

typedef __attribute__((ext_vector_type(16))) __bf16 v16bf;
typedef __attribute__((ext_vector_type(8)))  __bf16 v8bf;
typedef __attribute__((ext_vector_type(8)))  float  v8f;
typedef int v4i_vs __attribute__((vector_size(16)));   // matches builtin param

#define HB_B 8
#define HB_T 2048
#define HB_D 512
#define HB_K 128
#define HB_L 6
#define HB_FRAME 512
#define HB_VOCAB 4096
#define HB_M (HB_B * HB_T)   // 16384 rows

#if defined(__has_builtin)
#if __has_builtin(__builtin_amdgcn_global_load_async_to_lds_b128) && \
    __has_builtin(__builtin_amdgcn_s_wait_asynccnt)
#define USE_ASYNC_COPY 1
#endif
#endif
#ifndef USE_ASYNC_COPY
#define USE_ASYNC_COPY 0
#endif

__device__ __forceinline__ float bf2f(__bf16 h) { return (float)h; }
__device__ __forceinline__ __bf16 f2bf(float f) { return (__bf16)f; }

__device__ __forceinline__ v16bf ld_frag(const __bf16* p0, const __bf16* p1) {
    union { v16bf v; v8bf h[2]; } u;
    u.h[0] = *(const v8bf*)p0;
    u.h[1] = *(const v8bf*)p1;
    return u.v;
}

#if USE_ASYNC_COPY
__device__ __forceinline__ void async_b128(const __bf16* g, __bf16* l) {
    __builtin_amdgcn_global_load_async_to_lds_b128(
        (__attribute__((address_space(1))) v4i_vs*)g,
        (__attribute__((address_space(3))) v4i_vs*)l,
        0, 0);
}
#endif

// ---------------------------------------------------------------------------
// GEMM: C[M,N] = A[M,K](bf16,row-major) * Bt[N,K](bf16,row-major)^T
//               (+ bias[N]) (+ residual[M,N] fp32)
// writes Cf (fp32) and/or Cb (bf16). M%128==0, N%128==0, K%32==0.
// Block: 128 threads = 4 waves (2x2); block tile 128x128; wave tile 64x64.
// Triple-buffered LDS, async global->LDS staging, one barrier per K-step.
// ---------------------------------------------------------------------------
#define GSA 40   // padded LDS row stride (elements) for 32-wide K tiles
#define GBK 32

// stage one 128x32 A tile + 128x32 B tile into LDS buffer `buf`
__device__ __forceinline__ void gemm_stage(
    const __bf16* __restrict__ A, int lda,
    const __bf16* __restrict__ Bt, int ldb,
    int m0, int n0, int k0,
    __bf16* as, __bf16* bs, int tid)
{
    #pragma unroll
    for (int i = 0; i < 8; ++i) {
        const int c   = tid + 128 * i;      // 16B chunk id, 0..1023
        const int row = c >> 2;             // 0..255 (A rows then B rows)
        const int off = (c & 3) * 8;        // element offset within 32-wide row
        const bool isA = row < 128;
        const int r = isA ? row : row - 128;
        const __bf16* g = (isA ? A + (size_t)(m0 + r) * lda
                               : Bt + (size_t)(n0 + r) * ldb) + k0 + off;
        __bf16* l = (isA ? as : bs) + r * GSA + off;
#if USE_ASYNC_COPY
        async_b128(g, l);                    // ASYNCcnt++ (8 per wave per tile)
#else
        *(v8bf*)l = *(const v8bf*)g;
#endif
    }
}

__global__ __launch_bounds__(128) void gemm_bf16_wmma(
    const __bf16* __restrict__ A, int lda,
    const __bf16* __restrict__ Bt, int ldb,
    const float* __restrict__ bias,
    const float* __restrict__ resid, int ldr,
    float* __restrict__ Cf, __bf16* __restrict__ Cb, int ldc,
    int M, int N, int K)
{
    __shared__ __bf16 As[3][128 * GSA];     // 3 x 10 KB
    __shared__ __bf16 Bs[3][128 * GSA];     // 3 x 10 KB  (60 KB total)

    const int tid  = threadIdx.x;
    const int wave = tid >> 5;              // 0..3
    const int lane = tid & 31;
    const int m0 = blockIdx.y * 128;
    const int n0 = blockIdx.x * 128;
    const int wm = (wave >> 1) * 64;        // wave row offset (2 waves in M)
    const int wn = (wave & 1) * 64;         // wave col offset (2 waves in N)

    v8f acc[4][4];
    #pragma unroll
    for (int mt = 0; mt < 4; ++mt)
        #pragma unroll
        for (int nt = 0; nt < 4; ++nt)
            #pragma unroll
            for (int r = 0; r < 8; ++r) acc[mt][nt][r] = 0.0f;

    const int lhalf = lane >> 4;            // lane group 0/1
    const int l15   = lane & 15;
    const int ka = lhalf * 8;               // A frag K base per lane group
    const int kb = lhalf * 8;               // B frag K base per lane group

    const int S = K / GBK;                  // K-steps (16 for K=512)

    // prologue: tiles 0 and 1 in flight
    gemm_stage(A, lda, Bt, ldb, m0, n0, 0,       As[0], Bs[0], tid);
    gemm_stage(A, lda, Bt, ldb, m0, n0, GBK,     As[1], Bs[1], tid);

    for (int k = 0; k < S; ++k) {
        const int buf = k % 3;
#if USE_ASYNC_COPY
        // own tile-k asyncs complete (tile k+1 = 8 newer may remain in flight)
        if (k + 1 < S) __builtin_amdgcn_s_wait_asynccnt(8);
        else           __builtin_amdgcn_s_wait_asynccnt(0);
#endif
        __syncthreads();                    // tile k visible to all waves
        // issue tile k+2 into buffer (k+2)%3 == (k-1)%3: safe, since every
        // wave finished compute(k-1) before passing the barrier above.
        if (k + 2 < S)
            gemm_stage(A, lda, Bt, ldb, m0, n0, (k + 2) * GBK,
                       As[(k + 2) % 3], Bs[(k + 2) % 3], tid);

        // B fragments: lane = column n (l15), K chunks [kb..kb+7],[kb+16..kb+23].
        v16bf bfr[4];
        #pragma unroll
        for (int nt = 0; nt < 4; ++nt) {
            const __bf16* bp = &Bs[buf][(size_t)(wn + nt * 16 + l15) * GSA + kb];
            bfr[nt] = ld_frag(bp, bp + 16);
        }
        // A fragments: lane = row m (l15), K chunks [ka..ka+7],[ka+16..ka+23].
        #pragma unroll
        for (int mt = 0; mt < 4; ++mt) {
            const __bf16* ap = &As[buf][(size_t)(wm + mt * 16 + l15) * GSA + ka];
            v16bf afr = ld_frag(ap, ap + 16);
            #pragma unroll
            for (int nt = 0; nt < 4; ++nt) {
                acc[mt][nt] = __builtin_amdgcn_wmma_f32_16x16x32_bf16(
                    false, afr, false, bfr[nt], (short)0, acc[mt][nt],
                    false, false);
            }
        }
    }

    // Epilogue. C/D layout: VGPR r -> M = r + (lane>=16)*8, N = lane&15.
    #pragma unroll
    for (int mt = 0; mt < 4; ++mt)
        #pragma unroll
        for (int nt = 0; nt < 4; ++nt)
            #pragma unroll
            for (int r = 0; r < 8; ++r) {
                int row = m0 + wm + mt * 16 + lhalf * 8 + r;
                int col = n0 + wn + nt * 16 + l15;
                float v = acc[mt][nt][r];
                if (bias)  v += bias[col];
                if (resid) v += resid[(size_t)row * ldr + col];
                if (Cf) Cf[(size_t)row * ldc + col] = v;
                if (Cb) Cb[(size_t)row * ldc + col] = f2bf(v);
            }
}

// ---------------------------------------------------------------------------
// LayerNorm over rows of 512 fp32 -> bf16. One wave per row, 8 rows/block.
// ---------------------------------------------------------------------------
__global__ __launch_bounds__(256) void layernorm_kernel(
    const float* __restrict__ x, const float* __restrict__ g,
    const float* __restrict__ bta, __bf16* __restrict__ out)
{
    const int row  = blockIdx.x * 8 + (threadIdx.x >> 5);
    const int lane = threadIdx.x & 31;
    const float* xr = x + (size_t)row * HB_D;

    float v[16];
    float s = 0.f, ss = 0.f;
    #pragma unroll
    for (int i = 0; i < 4; ++i) {
        float4 q = ((const float4*)xr)[lane + 32 * i];
        v[4*i+0] = q.x; v[4*i+1] = q.y; v[4*i+2] = q.z; v[4*i+3] = q.w;
    }
    #pragma unroll
    for (int i = 0; i < 16; ++i) { s += v[i]; ss += v[i] * v[i]; }
    #pragma unroll
    for (int off = 16; off >= 1; off >>= 1) {
        s  += __shfl_xor(s,  off, 32);
        ss += __shfl_xor(ss, off, 32);
    }
    const float mu  = s * (1.0f / HB_D);
    const float var = ss * (1.0f / HB_D) - mu * mu;
    const float rs  = rsqrtf(var + 1e-5f);

    __bf16* orow = out + (size_t)row * HB_D;
    #pragma unroll
    for (int i = 0; i < 4; ++i) {
        const int base = (lane + 32 * i) * 4;
        #pragma unroll
        for (int c = 0; c < 4; ++c) {
            const int idx = base + c;
            orow[idx] = f2bf((v[4*i+c] - mu) * rs * g[idx] + bta[idx]);
        }
    }
}

// ---------------------------------------------------------------------------
// Causal depthwise conv (128 taps) + gating:
//   yout[b,t,d] = ( sum_{j=0..127} ker[d,j] * yin[b,t-j,d] ) * xi[b,t,d]
// Block: 32 channels x 64 timesteps; fp32 accumulate in registers.
// ---------------------------------------------------------------------------
#define CY_S 200   // padded y-window row stride (192 live)
#define CK_S 132   // padded tap row stride (128 live)

__global__ __launch_bounds__(256) void hyena_conv(
    const __bf16* __restrict__ yin, int ldy,
    const __bf16* __restrict__ xi,  int ldxi,
    const float* __restrict__ ker,           // [512][128]
    __bf16* __restrict__ yout)               // ld = 512
{
    __shared__ __bf16 Ys[32 * CY_S];
    __shared__ float  Ks[32 * CK_S];

    const int b  = blockIdx.z;
    const int d0 = blockIdx.y * 32;
    const int t0 = blockIdx.x * 64;
    const int tid = threadIdx.x;

    // y window: global t in [t0-128, t0+63]  -> Ys[d][0..191]
    for (int idx = tid; idx < 192 * 32; idx += 256) {
        const int tt = idx >> 5;
        const int dd = idx & 31;
        const int t  = t0 - 128 + tt;
        __bf16 val = f2bf(0.0f);
        if (t >= 0) val = yin[((size_t)b * HB_T + t) * ldy + d0 + dd];
        Ys[dd * CY_S + tt] = val;
    }
    for (int idx = tid; idx < 32 * 128; idx += 256) {
        const int dd = idx >> 7;
        const int jj = idx & 127;
        Ks[dd * CK_S + jj] = ker[(size_t)(d0 + dd) * HB_K + jj];
    }
    __syncthreads();

    const int dd = tid & 31;
    const int tb = (tid >> 5) * 8;      // outputs t0+tb .. t0+tb+7
    const __bf16* yrow = &Ys[dd * CY_S];

    float acc[8];
    #pragma unroll
    for (int i = 0; i < 8; ++i) acc[i] = 0.0f;

    for (int c = 0; c < 16; ++c) {      // 8 taps per chunk
        const int base = tb + 120 - 8 * c;        // multiple of 8 -> 16B aligned
        v8bf w0 = *(const v8bf*)(yrow + base);
        v8bf w1 = *(const v8bf*)(yrow + base + 8);
        float yw[16];
        #pragma unroll
        for (int m = 0; m < 8; ++m) { yw[m] = bf2f(w0[m]); yw[8 + m] = bf2f(w1[m]); }
        const float* kp = &Ks[dd * CK_S + 8 * c];
        float kk[8];
        #pragma unroll
        for (int m = 0; m < 8; ++m) kk[m] = kp[m];
        #pragma unroll
        for (int jj = 0; jj < 8; ++jj)
            #pragma unroll
            for (int i = 0; i < 8; ++i)
                acc[i] += kk[jj] * yw[8 + i - jj];   // y[t0+tb+i-(8c+jj)]
    }

    #pragma unroll
    for (int i = 0; i < 8; ++i) {
        const int t = t0 + tb + i;
        const float xv = bf2f(xi[((size_t)b * HB_T + t) * ldxi + d0 + dd]);
        yout[((size_t)b * HB_T + t) * HB_D + d0 + dd] = f2bf(acc[i] * xv);
    }
}

// ---------------------------------------------------------------------------
// fp32 -> bf16 elementwise, and weight convert+transpose [K][N] -> [N][K].
// ---------------------------------------------------------------------------
__global__ void cvt_bf16_kernel(const float* __restrict__ x,
                                __bf16* __restrict__ y, int n)
{
    int idx = blockIdx.x * 256 + threadIdx.x;
    if (idx < n) y[idx] = f2bf(x[idx]);
}

__global__ void wconvt_kernel(const float* __restrict__ W,
                              __bf16* __restrict__ Wt, int K, int N)
{
    int idx = blockIdx.x * 256 + threadIdx.x;
    if (idx < K * N) {
        int k = idx / N;
        int n = idx - k * N;
        Wt[(size_t)n * K + k] = f2bf(W[idx]);
    }
}

// ---------------------------------------------------------------------------
// Host orchestration.
// ---------------------------------------------------------------------------
extern "C" void kernel_launch(void* const* d_in, const int* in_sizes, int n_in,
                              void* d_out, int out_size, void* d_ws, size_t ws_size,
                              hipStream_t stream) {
    (void)in_sizes; (void)n_in; (void)out_size; (void)ws_size;

    const float* frames  = (const float*)d_in[0];
    const float* frame_W = (const float*)d_in[1];
    const float* frame_b = (const float*)d_in[2];
    const float* ln_g    = (const float*)d_in[3];
    const float* ln_b    = (const float*)d_in[4];
    const float* in_W    = (const float*)d_in[5];
    const float* in_b    = (const float*)d_in[6];
    const float* kernels = (const float*)d_in[7];
    const float* out_W   = (const float*)d_in[8];
    const float* out_b   = (const float*)d_in[9];
    const float* fln_g   = (const float*)d_in[10];
    const float* fln_b   = (const float*)d_in[11];
    const float* sym_W   = (const float*)d_in[12];
    const float* sym_b   = (const float*)d_in[13];
    const float* rec_W   = (const float*)d_in[14];
    const float* rec_b   = (const float*)d_in[15];

    const size_t M = HB_M;

    // workspace carve-up
    char* w = (char*)d_ws;
    float*  x_f  = (float*)w;              w += M * HB_D * 4;          // fp32 residual stream
    __bf16* xn   = (__bf16*)w;             w += M * HB_D * 2;          // LN out (also reused as `hid`)
    __bf16* proj = (__bf16*)w;             w += M * (3 * HB_D) * 2;    // 48 MB
    __bf16* ya   = (__bf16*)w;             w += M * HB_D * 2;          // conv ping (also frames_bf16)
    __bf16* yb   = (__bf16*)w;             w += M * HB_D * 2;          // conv pong
    __bf16* frame_Wt = (__bf16*)w;         w += (size_t)HB_D * HB_FRAME * 2;
    __bf16* in_Wt    = (__bf16*)w;         w += (size_t)HB_L * 3 * HB_D * HB_D * 2;
    __bf16* out_Wt   = (__bf16*)w;         w += (size_t)HB_L * HB_D * HB_D * 2;
    __bf16* sym_Wt   = (__bf16*)w;         w += (size_t)HB_VOCAB * HB_D * 2;
    __bf16* rec_Wt   = (__bf16*)w;         w += (size_t)HB_FRAME * HB_D * 2;

    __bf16* frames_bf = ya;   // alias: only needed before the layer loop
    __bf16* hid       = xn;   // alias: only needed after the layer loop

    // --- weight conversion + transpose (once per launch) ---
    {
        int n = (int)(M * HB_FRAME);
        cvt_bf16_kernel<<<(n + 255) / 256, 256, 0, stream>>>(frames, frames_bf, n);
    }
    wconvt_kernel<<<(HB_FRAME * HB_D + 255) / 256, 256, 0, stream>>>(frame_W, frame_Wt, HB_FRAME, HB_D);
    for (int l = 0; l < HB_L; ++l) {
        wconvt_kernel<<<(HB_D * 3 * HB_D + 255) / 256, 256, 0, stream>>>(
            in_W + (size_t)l * HB_D * 3 * HB_D, in_Wt + (size_t)l * 3 * HB_D * HB_D, HB_D, 3 * HB_D);
        wconvt_kernel<<<(HB_D * HB_D + 255) / 256, 256, 0, stream>>>(
            out_W + (size_t)l * HB_D * HB_D, out_Wt + (size_t)l * HB_D * HB_D, HB_D, HB_D);
    }
    wconvt_kernel<<<(HB_D * HB_VOCAB + 255) / 256, 256, 0, stream>>>(sym_W, sym_Wt, HB_D, HB_VOCAB);
    wconvt_kernel<<<(HB_D * HB_FRAME + 255) / 256, 256, 0, stream>>>(rec_W, rec_Wt, HB_D, HB_FRAME);

    // --- front projection: x = frames @ frame_W + frame_b ---
    {
        dim3 grid(HB_D / 128, (unsigned)(M / 128));
        gemm_bf16_wmma<<<grid, 128, 0, stream>>>(
            frames_bf, HB_FRAME, frame_Wt, HB_FRAME,
            frame_b, nullptr, 0, x_f, nullptr, HB_D,
            (int)M, HB_D, HB_FRAME);
    }

    // --- Hyena blocks ---
    for (int l = 0; l < HB_L; ++l) {
        layernorm_kernel<<<(unsigned)(M / 8), 256, 0, stream>>>(
            x_f, ln_g + (size_t)l * HB_D, ln_b + (size_t)l * HB_D, xn);

        {   // proj = xn @ in_W[l] + in_b[l]   -> bf16 [M,1536]
            dim3 grid(3 * HB_D / 128, (unsigned)(M / 128));
            gemm_bf16_wmma<<<grid, 128, 0, stream>>>(
                xn, HB_D, in_Wt + (size_t)l * 3 * HB_D * HB_D, HB_D,
                in_b + (size_t)l * 3 * HB_D, nullptr, 0,
                nullptr, proj, 3 * HB_D,
                (int)M, 3 * HB_D, HB_D);
        }

        dim3 cgrid(HB_T / 64, HB_D / 32, HB_B);
        // stage 0: y = conv(proj[:,:,0,:], ker[l,0]) * proj[:,:,1,:]
        hyena_conv<<<cgrid, 256, 0, stream>>>(
            proj, 3 * HB_D, proj + HB_D, 3 * HB_D,
            kernels + ((size_t)l * 2 + 0) * HB_D * HB_K, ya);
        // stage 1: y = conv(y, ker[l,1]) * proj[:,:,2,:]
        hyena_conv<<<cgrid, 256, 0, stream>>>(
            ya, HB_D, proj + 2 * HB_D, 3 * HB_D,
            kernels + ((size_t)l * 2 + 1) * HB_D * HB_K, yb);

        {   // x = x + y @ out_W[l] + out_b[l]
            dim3 grid(HB_D / 128, (unsigned)(M / 128));
            gemm_bf16_wmma<<<grid, 128, 0, stream>>>(
                yb, HB_D, out_Wt + (size_t)l * HB_D * HB_D, HB_D,
                out_b + (size_t)l * HB_D, x_f, HB_D,
                x_f, nullptr, HB_D,
                (int)M, HB_D, HB_D);
        }
    }

    // --- final LN + output heads ---
    layernorm_kernel<<<(unsigned)(M / 8), 256, 0, stream>>>(x_f, fln_g, fln_b, hid);

    float* out_frames = (float*)d_out;                 // [M,512]
    float* out_logits = out_frames + M * HB_FRAME;     // [M,4096]
    {
        dim3 grid(HB_FRAME / 128, (unsigned)(M / 128));
        gemm_bf16_wmma<<<grid, 128, 0, stream>>>(
            hid, HB_D, rec_Wt, HB_D, rec_b, nullptr, 0,
            out_frames, nullptr, HB_FRAME, (int)M, HB_FRAME, HB_D);
    }
    {
        dim3 grid(HB_VOCAB / 128, (unsigned)(M / 128));
        gemm_bf16_wmma<<<grid, 128, 0, stream>>>(
            hid, HB_D, sym_Wt, HB_D, sym_b, nullptr, 0,
            out_logits, nullptr, HB_VOCAB, (int)M, HB_VOCAB, HB_D);
    }
}
